// AttentionLayer_7602092113932
// MI455X (gfx1250) — matrix operand order repaired and checked by reading, not compiled, not run on hardware
//
#include <hip/hip_runtime.h>
#include <hip/hip_bf16.h>
#include <math.h>

// Problem sizes (fixed by the reference)
#define BB  32
#define LL  2048
#define DD  512
#define DAA 512
#define NLL 512

typedef __attribute__((ext_vector_type(16))) __bf16 v16bf;
typedef __attribute__((ext_vector_type(8)))  __bf16 v8bf;
typedef __attribute__((ext_vector_type(8)))  float  v8f;

union BFrag {
    v16bf v;
    v8bf  h[2];
};

__device__ __forceinline__ v8f wmma_bf16(v16bf a, v16bf b, v8f c) {
    // D = A(16x32 bf16) * B(32x16 bf16) + C(16x16 f32)
    return __builtin_amdgcn_wmma_f32_16x16x32_bf16(
        /*neg_a=*/false, a, /*neg_b=*/false, b,
        /*c_mod=*/(short)0, c, /*reuse_a=*/false, /*reuse_b=*/false);
}

__device__ __forceinline__ v8f zero8() {
    v8f z = {0.f, 0.f, 0.f, 0.f, 0.f, 0.f, 0.f, 0.f};
    return z;
}

// Branch-free tanh using raw v_exp_f32 + v_rcp_f32 (no IEEE div sequence).
// tanh(x) = 1 - 2/(exp(2x)+1).  exp->inf => rcp=0 => 1;  exp->0 => 1-2 = -1.
// Error ~1e-6, far below the bf16 quantization applied right after.
__device__ __forceinline__ float fast_tanh(float v) {
    float e = __expf(2.f * v);
    return 1.f - 2.f * __builtin_amdgcn_rcpf(e + 1.f);
}

// Convert 8 contiguous f32 -> v8bf (two aligned float4 loads)
__device__ __forceinline__ v8bf cvt8(const float* __restrict__ p) {
    float4 a = *(const float4*)(p);
    float4 b = *(const float4*)(p + 4);
    v8bf r;
    r[0] = (__bf16)a.x; r[1] = (__bf16)a.y; r[2] = (__bf16)a.z; r[3] = (__bf16)a.w;
    r[4] = (__bf16)b.x; r[5] = (__bf16)b.y; r[6] = (__bf16)b.z; r[7] = (__bf16)b.w;
    return r;
}

// Load 2 chunks of 8 bf16 at p and p+16 (B-fragment for B = W^T with W row-major: row n of W)
__device__ __forceinline__ v16bf ldb16(const __bf16* __restrict__ p) {
    BFrag f;
    f.h[0] = *(const v8bf*)(p);
    f.h[1] = *(const v8bf*)(p + 16);
    return f.v;
}

// ---------------------------------------------------------------------------
// Kernel 1: convert W1, W2 (512x512 f32, row-major [N,K]) to bf16 in ws
// ---------------------------------------------------------------------------
__global__ __launch_bounds__(256) void prep_weights(
    const float* __restrict__ W1, const float* __restrict__ W2,
    __bf16* __restrict__ W1b, __bf16* __restrict__ W2b) {
    int i = blockIdx.x * 256 + threadIdx.x;
    if (i < DAA * DD) {
        W1b[i] = (__bf16)W1[i];
        W2b[i] = (__bf16)W2[i];
    }
}

// ---------------------------------------------------------------------------
// Kernel 2: fused  w = tanh(x @ W1^T)  (kept in LDS, bf16)
//           logits = w @ W2^T          stored TRANSPOSED -> attw [B][NL][L]
// One block = 16 rows of M=(b,l); 256 threads = 8 waves, 4 N-tiles each.
// ---------------------------------------------------------------------------
__global__ __launch_bounds__(256) void fused_g12(
    const float* __restrict__ x, const __bf16* __restrict__ W1b,
    const __bf16* __restrict__ W2b, float* __restrict__ attw) {
    constexpr int WROW = DAA + 8;              // 520 halves = 1040 B (16B aligned, bank-conflict-free)
    __shared__ __bf16 wlds[16 * WROW];         // 16.25 KB

    const int lane = threadIdx.x & 31;
    const int wave = threadIdx.x >> 5;
    const int m    = lane & 15;                // fragment row / column index
    const int h    = lane >> 4;                // half-select
    const int row0 = blockIdx.x * 16;          // flattened (b,l) row base
    const int b    = row0 >> 11;               // row0 / L
    const int l0   = row0 & (LL - 1);

    // -------- Phase A: 16 x 512 tile of w = tanh(x @ W1^T) --------
    const float* xrow = x + (size_t)(row0 + m) * DD;
    v8f acc[4] = {zero8(), zero8(), zero8(), zero8()};

    for (int ks = 0; ks < DD / 32; ++ks) {
        const int k0 = ks * 32;
        BFrag A;                               // A[m][k]: 2 chunks of 8 per lane
        A.h[0] = cvt8(xrow + k0 + 8 * h);
        A.h[1] = cvt8(xrow + k0 + 16 + 8 * h);
        v16bf Bf[4];
#pragma unroll
        for (int j = 0; j < 4; ++j) {          // preload all B frags (clause-able)
            const int n0 = (wave * 4 + j) * 16;
            Bf[j] = ldb16(W1b + (size_t)(n0 + m) * DD + k0 + 8 * h);
        }
#pragma unroll
        for (int j = 0; j < 4; ++j)
            acc[j] = wmma_bf16(A.v, Bf[j], acc[j]);
    }
#pragma unroll
    for (int j = 0; j < 4; ++j) {
        const int n0 = (wave * 4 + j) * 16;
#pragma unroll
        for (int r = 0; r < 8; ++r) {
            float v = fast_tanh(acc[j][r]);    // C[m=r+8h][n=n0+m]
            wlds[(r + 8 * h) * WROW + n0 + m] = (__bf16)v;
        }
    }
    __syncthreads();

    // -------- Phase B: logits tile = w_tile @ W2^T, store transposed --------
    v8f acc2[4] = {zero8(), zero8(), zero8(), zero8()};
    const __bf16* wrow = wlds + (size_t)m * WROW;

    for (int ks = 0; ks < DAA / 32; ++ks) {
        const int k0 = ks * 32;
        BFrag A;
        A.h[0] = *(const v8bf*)(wrow + k0 + 8 * h);
        A.h[1] = *(const v8bf*)(wrow + k0 + 16 + 8 * h);
        v16bf Bf[4];
#pragma unroll
        for (int j = 0; j < 4; ++j) {
            const int n0 = (wave * 4 + j) * 16;
            Bf[j] = ldb16(W2b + (size_t)(n0 + m) * DAA + k0 + 8 * h);
        }
#pragma unroll
        for (int j = 0; j < 4; ++j)
            acc2[j] = wmma_bf16(A.v, Bf[j], acc2[j]);
    }
#pragma unroll
    for (int j = 0; j < 4; ++j) {
        const int n0 = (wave * 4 + j) * 16;
        float* op = attw + ((size_t)(b * NLL + n0 + m)) * LL + l0 + 8 * h;
        float4 s0 = {acc2[j][0], acc2[j][1], acc2[j][2], acc2[j][3]};
        float4 s1 = {acc2[j][4], acc2[j][5], acc2[j][6], acc2[j][7]};
        *(float4*)(op)     = s0;               // 8 consecutive l per lane
        *(float4*)(op + 4) = s1;
    }
}

// ---------------------------------------------------------------------------
// Kernel 3: softmax over L per (b,n) row (rows are contiguous after transpose)
// in-place f32 in attw, plus bf16 copy into attb (ws)
// ---------------------------------------------------------------------------
__global__ __launch_bounds__(256) void softmax_k(
    float* __restrict__ attw, __bf16* __restrict__ attb) {
    const int row  = blockIdx.x;               // b*NL + n
    const int t    = threadIdx.x;
    const int lane = t & 31, wv = t >> 5;
    float* p = attw + (size_t)row * LL;

    float v[8];
    {
        float4 a = *(const float4*)(p + t * 8);
        float4 c = *(const float4*)(p + t * 8 + 4);
        v[0] = a.x; v[1] = a.y; v[2] = a.z; v[3] = a.w;
        v[4] = c.x; v[5] = c.y; v[6] = c.z; v[7] = c.w;
    }
    float mx = v[0];
#pragma unroll
    for (int i = 1; i < 8; ++i) mx = fmaxf(mx, v[i]);
#pragma unroll
    for (int off = 16; off > 0; off >>= 1) mx = fmaxf(mx, __shfl_xor(mx, off, 32));
    __shared__ float redm[8];
    if (lane == 0) redm[wv] = mx;
    __syncthreads();
    float m8 = redm[0];
#pragma unroll
    for (int i = 1; i < 8; ++i) m8 = fmaxf(m8, redm[i]);

    float s = 0.f;
#pragma unroll
    for (int i = 0; i < 8; ++i) { v[i] = __expf(v[i] - m8); s += v[i]; }
#pragma unroll
    for (int off = 16; off > 0; off >>= 1) s += __shfl_xor(s, off, 32);
    __shared__ float reds[8];
    if (lane == 0) reds[wv] = s;
    __syncthreads();
    float tot = 0.f;
#pragma unroll
    for (int i = 0; i < 8; ++i) tot += reds[i];
    const float inv = 1.f / tot;

    v8bf o;
#pragma unroll
    for (int i = 0; i < 8; ++i) { v[i] *= inv; o[i] = (__bf16)v[i]; }
    float4 s0 = {v[0], v[1], v[2], v[3]};
    float4 s1 = {v[4], v[5], v[6], v[7]};
    *(float4*)(p + t * 8)     = s0;
    *(float4*)(p + t * 8 + 4) = s1;
    *(v8bf*)(attb + (size_t)row * LL + t * 8) = o;
}

// ---------------------------------------------------------------------------
// Kernel 4: context[b][n][d] = sum_l att_t[b][n][l] * x[b][l][d]
// Computed as C'[m=d][n=label]; transposed store -> contiguous in d.
// Block: one b, 32-d tile, 128 labels (8 waves x 16 labels, 2 WMMA each).
// x staged via double-buffered LDS transpose (1 barrier per k-step);
// one B fragment (att row) feeds both d-halves.
// ---------------------------------------------------------------------------
__global__ __launch_bounds__(256) void g3_ctx(
    const float* __restrict__ x, const __bf16* __restrict__ attb,
    float* __restrict__ ctx) {
    constexpr int XROW = 40;                   // 32 + 8 halves (80 B row, conflict-free, 16B aligned)
    __shared__ __bf16 xT[2][32 * XROW];        // double-buffered, 2 x 2.5 KB

    const int lane = threadIdx.x & 31;
    const int wave = threadIdx.x >> 5;
    const int m    = lane & 15;
    const int h    = lane >> 4;

    const int blk = blockIdx.x;                // 32 b * 16 dtiles * 4 groups = 2048
    const int b   = blk >> 6;
    const int rem = blk & 63;
    const int d0  = (rem >> 2) * 32;
    const int n0  = (rem & 3) * 128 + wave * 16;

    const int t   = threadIdx.x;
    const int skk = t >> 3;                    // staged k row 0..31
    const int sdp = (t & 7) * 4;               // staged d quad 0..28

    const float*  xb = x + (size_t)b * LL * DD;
    const __bf16* ap = attb + ((size_t)(b * NLL + n0 + m)) * LL;

    v8f acc0 = zero8(), acc1 = zero8();
    for (int ks = 0; ks < LL / 32; ++ks) {
        const int k0 = ks * 32;
        __bf16* buf = xT[ks & 1];
        {   // stage x[b][k0..k0+31][d0..d0+31] transposed into LDS as bf16
            float4 xv = *(const float4*)(xb + (size_t)(k0 + skk) * DD + d0 + sdp);
            buf[(sdp + 0) * XROW + skk] = (__bf16)xv.x;
            buf[(sdp + 1) * XROW + skk] = (__bf16)xv.y;
            buf[(sdp + 2) * XROW + skk] = (__bf16)xv.z;
            buf[(sdp + 3) * XROW + skk] = (__bf16)xv.w;
        }
        __syncthreads();                       // stage->sync->compute; 2 buffers make this safe
        v16bf Bf = ldb16(ap + k0 + 8 * h);     // B[k][n=label]: contiguous att_t row, shared by both halves
        BFrag A0, A1;                          // A[m=d][k] from transposed LDS tile
        A0.h[0] = *(const v8bf*)(buf + m * XROW + 8 * h);
        A0.h[1] = *(const v8bf*)(buf + m * XROW + 16 + 8 * h);
        A1.h[0] = *(const v8bf*)(buf + (16 + m) * XROW + 8 * h);
        A1.h[1] = *(const v8bf*)(buf + (16 + m) * XROW + 16 + 8 * h);
        acc0 = wmma_bf16(A0.v, Bf, acc0);
        acc1 = wmma_bf16(A1.v, Bf, acc1);
    }
    // c[r] = C'[d = d0 (+16) + r + 8h][n = n0 + m]  ->  ctx[b][n][d] contiguous
    float* op = ctx + ((size_t)(b * NLL + n0 + m)) * DD + d0 + 8 * h;
    float4 a0 = {acc0[0], acc0[1], acc0[2], acc0[3]};
    float4 a1 = {acc0[4], acc0[5], acc0[6], acc0[7]};
    float4 b0 = {acc1[0], acc1[1], acc1[2], acc1[3]};
    float4 b1 = {acc1[4], acc1[5], acc1[6], acc1[7]};
    *(float4*)(op)          = a0;
    *(float4*)(op + 4)      = a1;
    *(float4*)(op + 16)     = b0;
    *(float4*)(op + 20)     = b1;
}

// ---------------------------------------------------------------------------
// Kernel 5: weighted_output[b][n] = <context[b][n][:], W3[n][:]> + b3[n]
// ---------------------------------------------------------------------------
__global__ __launch_bounds__(256) void wout_k(
    const float* __restrict__ ctx, const float* __restrict__ W3,
    const float* __restrict__ b3, float* __restrict__ wout) {
    const int row  = blockIdx.x;               // b*NL + n
    const int n    = row & (NLL - 1);
    const int t    = threadIdx.x;
    const int lane = t & 31, wv = t >> 5;
    const float* c = ctx + (size_t)row * DD;
    const float* w = W3 + (size_t)n * DD;
    float s = c[t] * w[t] + c[t + 256] * w[t + 256];
#pragma unroll
    for (int off = 16; off > 0; off >>= 1) s += __shfl_xor(s, off, 32);
    __shared__ float red[8];
    if (lane == 0) red[wv] = s;
    __syncthreads();
    if (t == 0) {
        float tot = 0.f;
#pragma unroll
        for (int i = 0; i < 8; ++i) tot += red[i];
        wout[row] = tot + b3[n];
    }
}

// ---------------------------------------------------------------------------
extern "C" void kernel_launch(void* const* d_in, const int* in_sizes, int n_in,
                              void* d_out, int out_size, void* d_ws, size_t ws_size,
                              hipStream_t stream) {
    const float* x  = (const float*)d_in[0];   // [B, L, D]
    const float* W1 = (const float*)d_in[1];   // [DA, D]
    const float* W2 = (const float*)d_in[2];   // [NL, DA]
    const float* W3 = (const float*)d_in[3];   // [NL, D]
    const float* b3 = (const float*)d_in[4];   // [NL]

    // d_out: context [B,NL,D] | weighted_output [B,NL] | att_weights [B,NL,L]
    float* ctx  = (float*)d_out;
    float* wout = ctx + (size_t)BB * NLL * DD;
    float* attw = wout + (size_t)BB * NLL;

    // workspace: W1b | W2b | att bf16 copy  (~68 MB)
    __bf16* W1b  = (__bf16*)d_ws;
    __bf16* W2b  = W1b + (size_t)DAA * DD;
    __bf16* attb = W2b + (size_t)NLL * DAA;

    prep_weights<<<(DAA * DD + 255) / 256, 256, 0, stream>>>(W1, W2, W1b, W2b);
    fused_g12<<<(BB * LL) / 16, 256, 0, stream>>>(x, W1b, W2b, attw);
    softmax_k<<<BB * NLL, 256, 0, stream>>>(attw, attb);
    g3_ctx<<<BB * 64, 256, 0, stream>>>(x, attb, ctx);
    wout_k<<<BB * NLL, 256, 0, stream>>>(ctx, W3, b3, wout);
}
